// ast_block_67697274520362
// MI455X (gfx1250) — matrix operand order repaired and checked by reading, not compiled, not run on hardware
//
#include <hip/hip_runtime.h>
#include <math.h>

typedef _Float16 h16;
typedef __attribute__((ext_vector_type(16))) _Float16 v16h;
typedef __attribute__((ext_vector_type(8)))  _Float16 v8h;
typedef __attribute__((ext_vector_type(8)))  float    v8f;

#define N_NODES   65536
#define NPG_      512
#define B_GRAPHS  128
#define E_EDGES   (N_NODES * 8)
#define ES_EDGES  (E_EDGES + N_NODES)   /* + self loops */
#define HID_      128
#define KTOP      256
#define NEG_SLOPE 0.2f

static inline int cdiv_i(long long a, int b) { return (int)((a + b - 1) / b); }

// ---------------- one-hot @ W_a1 == row gather of W_a1, cast to f16 ----------
__global__ __launch_bounds__(256) void k_embed(const int* __restrict__ x,
                                               const float* __restrict__ W,
                                               h16* __restrict__ z) {
    int t = blockIdx.x * 256 + threadIdx.x;
    if (t >= N_NODES * 160) return;
    int i = t / 160, j = t - i * 160;
    z[t] = (h16)W[x[i] * 160 + j];
}

// ---------------- W (f32 [K,M]) -> Wt (f16 [M,K]) ---------------------------
__global__ __launch_bounds__(256) void k_convert_w(const float* __restrict__ W,
                                                   h16* __restrict__ Wt,
                                                   int K, int M) {
    int t = blockIdx.x * 256 + threadIdx.x;
    if (t >= K * M) return;
    int k = t / M, m = t - k * M;
    Wt[m * K + k] = (h16)W[k * M + m];
}

// ---------------- f16 WMMA GEMM: C[N,M] = A[N,K] * Bt[M,K]^T ----------------
// one wave per 16x16 tile; no divergence -> EXEC all ones for WMMA.
__global__ __launch_bounds__(256) void k_gemm_wmma(const h16* __restrict__ A,
                                                   const h16* __restrict__ Bt,
                                                   h16* __restrict__ C,
                                                   int K, int M) {
    int wave = blockIdx.x * (blockDim.x >> 5) + (threadIdx.x >> 5);
    int lane = threadIdx.x & 31;
    int mtile = wave / (M >> 4);
    int ntile = wave - mtile * (M >> 4);
    int lo = lane & 15, hi = lane >> 4;

    const h16* arow = A  + (size_t)(mtile * 16 + lo) * K + 8  * hi;
    const h16* bcol = Bt + (size_t)(ntile * 16 + lo) * K + 8  * hi;

    v8f acc = {};
    for (int kb = 0; kb < K; kb += 32) {
        union { v16h v; v8h p[2]; } a, b;
        a.p[0] = *(const v8h*)(arow + kb);        // k = kb+8hi .. +7
        a.p[1] = *(const v8h*)(arow + kb + 16);   // k = kb+8hi+16 .. +23
        b.p[0] = *(const v8h*)(bcol + kb);        // k = kb+8hi .. +7
        b.p[1] = *(const v8h*)(bcol + kb + 16);   // k = kb+8hi+16 .. +23
        acc = __builtin_amdgcn_wmma_f32_16x16x32_f16(false, a.v, false, b.v,
                                                     (short)0, acc, false, false);
    }
    // C layout: VGPR r -> row r (lanes 0-15) / row 8+r (lanes 16-31), col = lane%16
    h16* crow = C + (size_t)(mtile * 16 + 8 * hi) * M + ntile * 16 + lo;
#pragma unroll
    for (int r = 0; r < 8; ++r) crow[(size_t)r * M] = (h16)acc[r];
}

// ---------------- per-node attention logits + init m/den --------------------
__global__ __launch_bounds__(256) void k_logits(const h16* __restrict__ z,
                                                const float* __restrict__ asrc,
                                                const float* __restrict__ adst,
                                                float* __restrict__ es,
                                                float* __restrict__ ed,
                                                float* __restrict__ m,
                                                float* __restrict__ den,
                                                int H, int F) {
    int t = blockIdx.x * 256 + threadIdx.x;
    if (t >= N_NODES * H) return;
    int i = t / H, h = t - i * H;
    const h16* zp = z + (size_t)i * (H * F) + h * F;
    const float* as = asrc + h * F;
    const float* ad = adst + h * F;
    float s0 = 0.f, s1 = 0.f;
    for (int f = 0; f < F; ++f) {
        float zv = (float)zp[f];
        s0 += zv * as[f];
        s1 += zv * ad[f];
    }
    es[t] = s0; ed[t] = s1;
    m[t] = -1e30f; den[t] = 0.f;
}

__global__ __launch_bounds__(256) void k_zero(float* __restrict__ p, int n) {
    int t = blockIdx.x * 256 + threadIdx.x;
    if (t < n) p[t] = 0.f;
}

__device__ __forceinline__ void atomicMaxF(float* a, float v) {
    if (v >= 0.f) atomicMax((int*)a, __float_as_int(v));
    else          atomicMin((unsigned int*)a, (unsigned int)__float_as_int(v));
}

__device__ __forceinline__ void edge_ends(int eid, const int* src, const int* dst,
                                          int& s, int& d) {
    if (eid < E_EDGES) { s = src[eid]; d = dst[eid]; }
    else               { s = d = eid - E_EDGES; }      // self loop
}

// ---------------- edge pass 1: segment max ----------------------------------
__global__ __launch_bounds__(256) void k_edge_max(const int* __restrict__ src,
                                                  const int* __restrict__ dst,
                                                  const float* __restrict__ es,
                                                  const float* __restrict__ ed,
                                                  float* __restrict__ m, int H) {
    int t = blockIdx.x * 256 + threadIdx.x;
    if (t >= ES_EDGES * H) return;
    int eid = t / H, h = t - eid * H;
    int s, d; edge_ends(eid, src, dst, s, d);
    float v = es[s * H + h] + ed[d * H + h];
    v = v > 0.f ? v : NEG_SLOPE * v;
    atomicMaxF(&m[d * H + h], v);
}

// ---------------- edge pass 2: segment sum of exp ---------------------------
__global__ __launch_bounds__(256) void k_edge_den(const int* __restrict__ src,
                                                  const int* __restrict__ dst,
                                                  const float* __restrict__ es,
                                                  const float* __restrict__ ed,
                                                  const float* __restrict__ m,
                                                  float* __restrict__ den, int H) {
    int t = blockIdx.x * 256 + threadIdx.x;
    if (t >= ES_EDGES * H) return;
    int eid = t / H, h = t - eid * H;
    int s, d; edge_ends(eid, src, dst, s, d);
    float v = es[s * H + h] + ed[d * H + h];
    v = v > 0.f ? v : NEG_SLOPE * v;
    atomicAdd(&den[d * H + h], expf(v - m[d * H + h]));
}

// ---------------- edge pass 3: weighted scatter of z[src] -------------------
__global__ __launch_bounds__(256) void k_edge_aggr(const int* __restrict__ src,
                                                   const int* __restrict__ dst,
                                                   const float* __restrict__ es,
                                                   const float* __restrict__ ed,
                                                   const float* __restrict__ m,
                                                   const float* __restrict__ den,
                                                   const h16* __restrict__ z,
                                                   float* __restrict__ acc,
                                                   int H, int F, int nch) {
    int t = blockIdx.x * 256 + threadIdx.x;
    if (t >= ES_EDGES * H * nch) return;
    int c = t % nch;
    int r = t / nch;
    int h = r % H;
    int eid = r / H;
    int s, d; edge_ends(eid, src, dst, s, d);
    float v = es[s * H + h] + ed[d * H + h];
    v = v > 0.f ? v : NEG_SLOPE * v;
    float alpha = expf(v - m[d * H + h]) / (den[d * H + h] + 1e-16f);
    int cs = (F + nch - 1) / nch;
    int f0 = c * cs, f1 = f0 + cs; if (f1 > F) f1 = F;
    const h16* zp = z + (size_t)s * (H * F) + h * F;
    float* ap = acc + (size_t)d * (H * F) + h * F;
    for (int f = f0; f < f1; ++f) atomicAdd(&ap[f], (float)zp[f] * alpha);
}

// ---------------- bias + relu ; keep f32 and emit f16 for next GEMM ---------
__global__ __launch_bounds__(256) void k_finalize(float* __restrict__ acc,
                                                  const float* __restrict__ bias,
                                                  h16* __restrict__ out, int M) {
    int t = blockIdx.x * 256 + threadIdx.x;
    if (t >= N_NODES * M) return;
    int j = t % M;
    float v = acc[t] + bias[j];
    v = v > 0.f ? v : 0.f;
    acc[t] = v;
    out[t] = (h16)v;
}

// ---------------- SAGPool: per-node projections -----------------------------
__global__ __launch_bounds__(256) void k_pq(const float* __restrict__ h32,
                                            const float* __restrict__ wrel,
                                            const float* __restrict__ wroot,
                                            float* __restrict__ p,
                                            float* __restrict__ q,
                                            float* __restrict__ s) {
    int i = blockIdx.x * 256 + threadIdx.x;
    if (i >= N_NODES) return;
    const float* hp = h32 + (size_t)i * HID_;
    float a = 0.f, b = 0.f;
    for (int f = 0; f < HID_; ++f) { a += hp[f] * wrel[f]; b += hp[f] * wroot[f]; }
    p[i] = a; q[i] = b; s[i] = 0.f;
}

__global__ __launch_bounds__(256) void k_pool_edge(const int* __restrict__ src,
                                                   const int* __restrict__ dst,
                                                   const float* __restrict__ p,
                                                   float* __restrict__ s) {
    int e = blockIdx.x * 256 + threadIdx.x;
    if (e >= E_EDGES) return;
    atomicAdd(&s[dst[e]], p[src[e]]);
}

__global__ __launch_bounds__(256) void k_score(const float* __restrict__ s,
                                               const float* __restrict__ q,
                                               const float* __restrict__ bp,
                                               float* __restrict__ score) {
    int i = blockIdx.x * 256 + threadIdx.x;
    if (i >= N_NODES) return;
    score[i] = tanhf(s[i] + q[i] + bp[0]);
}

// ---------------- per-graph bitonic top-K + max pool ------------------------
__global__ __launch_bounds__(256) void k_topk_pool(const float* __restrict__ score,
                                                   const float* __restrict__ h32,
                                                   float* __restrict__ pooled,
                                                   float* __restrict__ perm,
                                                   float* __restrict__ sv,
                                                   float* __restrict__ bout) {
    __shared__ float ssc[NPG_];
    __shared__ int   sid[NPG_];
    int g = blockIdx.x, tid = threadIdx.x;
    for (int t = tid; t < NPG_; t += 256) { ssc[t] = score[g * NPG_ + t]; sid[t] = t; }
    __syncthreads();
    // bitonic sort, descending by score, tie-break lower index first
    for (int k = 2; k <= NPG_; k <<= 1) {
        for (int j = k >> 1; j > 0; j >>= 1) {
            for (int t = tid; t < NPG_; t += 256) {
                int x = t ^ j;
                if (x > t) {
                    bool up = ((t & k) == 0);          // descending segment
                    float s1 = ssc[t], s2 = ssc[x];
                    int   i1 = sid[t], i2 = sid[x];
                    bool keep = (s1 > s2) || (s1 == s2 && i1 < i2);
                    if (up != keep) { ssc[t] = s2; ssc[x] = s1; sid[t] = i2; sid[x] = i1; }
                }
            }
            __syncthreads();
        }
    }
    if (tid < KTOP) {
        perm[g * KTOP + tid] = (float)(g * NPG_ + sid[tid]);
        sv[g * KTOP + tid]   = ssc[tid];
        bout[g * KTOP + tid] = (float)g;
    }
    __syncthreads();
    if (tid < HID_) {
        float mx = -1e30f;
        for (int t = 0; t < KTOP; ++t) {
            int ng = g * NPG_ + sid[t];
            mx = fmaxf(mx, h32[(size_t)ng * HID_ + tid] * ssc[t]);
        }
        pooled[g * HID_ + tid] = mx;
    }
}

// ============================================================================
extern "C" void kernel_launch(void* const* d_in, const int* in_sizes, int n_in,
                              void* d_out, int out_size, void* d_ws, size_t ws_size,
                              hipStream_t stream) {
    (void)in_sizes; (void)n_in; (void)out_size; (void)ws_size;
    const int*   x    = (const int*)d_in[0];
    const int*   src  = (const int*)d_in[1];
    const int*   dst  = src + E_EDGES;
    const float* W_a1 = (const float*)d_in[3];
    const float* as_a1 = (const float*)d_in[4];
    const float* ad_a1 = (const float*)d_in[5];
    const float* b_a1  = (const float*)d_in[6];
    const float* W_a2  = (const float*)d_in[7];
    const float* as_a2 = (const float*)d_in[8];
    const float* ad_a2 = (const float*)d_in[9];
    const float* b_a2  = (const float*)d_in[10];
    const float* W_b1  = (const float*)d_in[11];
    const float* as_b1 = (const float*)d_in[12];
    const float* ad_b1 = (const float*)d_in[13];
    const float* b_b1  = (const float*)d_in[14];
    const float* W_b2  = (const float*)d_in[15];
    const float* as_b2 = (const float*)d_in[16];
    const float* ad_b2 = (const float*)d_in[17];
    const float* b_b2  = (const float*)d_in[18];
    const float* w_rel  = (const float*)d_in[19];
    const float* w_root = (const float*)d_in[20];
    const float* b_pool = (const float*)d_in[21];

    // --- workspace carve-up (256B aligned slices) ---
    size_t off = 0;
    auto carve = [&](size_t bytes) -> char* {
        char* p = (char*)d_ws + off;
        off += (bytes + 255) & ~(size_t)255;
        return p;
    };
    h16*   bufA = (h16*)  carve((size_t)N_NODES * 160 * sizeof(h16));  // layer input f16
    h16*   bufB = (h16*)  carve((size_t)N_NODES * 160 * sizeof(h16));  // GEMM / embed out z
    float* acc  = (float*)carve((size_t)N_NODES * 160 * sizeof(float));
    float* es   = (float*)carve((size_t)N_NODES * 8   * sizeof(float));
    float* ed   = (float*)carve((size_t)N_NODES * 8   * sizeof(float));
    float* mm   = (float*)carve((size_t)N_NODES * 8   * sizeof(float));
    float* den  = (float*)carve((size_t)N_NODES * 8   * sizeof(float));
    h16*   Wt   = (h16*)  carve((size_t)160 * 160 * sizeof(h16));
    float* pv   = (float*)carve((size_t)N_NODES * sizeof(float));
    float* qv   = (float*)carve((size_t)N_NODES * sizeof(float));
    float* sv_s = (float*)carve((size_t)N_NODES * sizeof(float));
    float* scr  = (float*)carve((size_t)N_NODES * sizeof(float));

    auto run_attention = [&](const h16* z, const float* asrc, const float* adst,
                             const float* bias, int H, int F, int nch) {
        int M = H * F;
        k_logits<<<cdiv_i((long long)N_NODES * H, 256), 256, 0, stream>>>(
            z, asrc, adst, es, ed, mm, den, H, F);
        k_zero<<<cdiv_i((long long)N_NODES * M, 256), 256, 0, stream>>>(acc, N_NODES * M);
        k_edge_max<<<cdiv_i((long long)ES_EDGES * H, 256), 256, 0, stream>>>(
            src, dst, es, ed, mm, H);
        k_edge_den<<<cdiv_i((long long)ES_EDGES * H, 256), 256, 0, stream>>>(
            src, dst, es, ed, mm, den, H);
        k_edge_aggr<<<cdiv_i((long long)ES_EDGES * H * nch, 256), 256, 0, stream>>>(
            src, dst, es, ed, mm, den, z, acc, H, F, nch);
        k_finalize<<<cdiv_i((long long)N_NODES * M, 256), 256, 0, stream>>>(
            acc, bias, bufA, M);
    };

    auto run_gemm = [&](const float* W, int K, int M) {
        k_convert_w<<<cdiv_i((long long)K * M, 256), 256, 0, stream>>>(W, Wt, K, M);
        int waves = (N_NODES / 16) * (M / 16);
        k_gemm_wmma<<<waves / 8, 256, 0, stream>>>(bufA, Wt, bufB, K, M);
    };

    // ---- GATNet A ----
    k_embed<<<cdiv_i((long long)N_NODES * 160, 256), 256, 0, stream>>>(x, W_a1, bufB);
    run_attention(bufB, as_a1, ad_a1, b_a1, 8, 20, 4);          // conv a1 + relu
    run_gemm(W_a2, 160, 128);                                   // z = h @ W_a2 (WMMA)
    run_attention(bufB, as_a2, ad_a2, b_a2, 1, 128, 8);         // conv a2 + outer relu
    // ---- GATNet B ----
    run_gemm(W_b1, 128, 160);
    run_attention(bufB, as_b1, ad_b1, b_b1, 8, 20, 4);          // conv b1 + relu
    run_gemm(W_b2, 160, 128);
    run_attention(bufB, as_b2, ad_b2, b_b2, 1, 128, 8);         // conv b2 + outer relu
    // acc now holds final h in f32 [N, 128]

    // ---- SAGPooling ----
    k_pq<<<cdiv_i(N_NODES, 256), 256, 0, stream>>>(acc, w_rel, w_root, pv, qv, sv_s);
    k_pool_edge<<<cdiv_i(E_EDGES, 256), 256, 0, stream>>>(src, dst, pv, sv_s);
    k_score<<<cdiv_i(N_NODES, 256), 256, 0, stream>>>(sv_s, qv, b_pool, scr);

    float* out_pooled = (float*)d_out;                           // [B, HID]
    float* out_perm   = out_pooled + B_GRAPHS * HID_;            // [B*K]
    float* out_sv     = out_perm   + B_GRAPHS * KTOP;            // [B*K]
    float* out_batch  = out_sv     + B_GRAPHS * KTOP;            // [B*K]
    k_topk_pool<<<B_GRAPHS, 256, 0, stream>>>(scr, acc, out_pooled, out_perm,
                                              out_sv, out_batch);
}
